// ScoreNet_78864189489250
// MI455X (gfx1250) — matrix operand order repaired and checked by reading, not compiled, not run on hardware
//
#include <hip/hip_runtime.h>
#include <hip/hip_bf16.h>

#define NNODE 4096
#define DMODEL 512
#define QKVD  1536
#define NHEAD 8
#define HDIM 64
#define CDIM 224   // concat dim padded to multiple of 32
#define CREAL 208  // real concat dim: 128 + 32 + 32 + 16

typedef __attribute__((ext_vector_type(16))) __bf16 v16bf;
typedef __attribute__((ext_vector_type(8)))  float  v8f;
typedef __attribute__((ext_vector_type(8)))  unsigned short u16x8;

union FragU {
  v16bf v;
  unsigned short h[16];
  u16x8 q[2];
};

// fp32 -> bf16, round-to-nearest-even, NaN/Inf passthrough
__device__ __forceinline__ unsigned short f2bf(float f) {
  union { float f; unsigned int u; } x;
  x.f = f;
  unsigned int u = x.u;
  unsigned int r = u + 0x7FFFu + ((u >> 16) & 1u);
  if ((u & 0x7F800000u) == 0x7F800000u) r = u;
  return (unsigned short)(r >> 16);
}

__device__ __forceinline__ v8f wmma_bf16(const FragU& a, const FragU& b, v8f c) {
  return __builtin_amdgcn_wmma_f32_16x16x32_bf16(false, a.v, false, b.v,
                                                 (short)0, c, false, false);
}

// ---------------- elementwise fp32 -> bf16 ----------------
__global__ void conv_bf16_kernel(const float* __restrict__ in,
                                 unsigned short* __restrict__ out, int n) {
  const int i = blockIdx.x * blockDim.x + threadIdx.x;
  if (i < n) out[i] = f2bf(in[i]);
}

// pad rows Kin -> Kout with zeros while converting (for proj_w 208 -> 224)
__global__ void conv_pad_kernel(const float* __restrict__ in,
                                unsigned short* __restrict__ out,
                                int rows, int Kin, int Kout) {
  const int i = blockIdx.x * blockDim.x + threadIdx.x;
  if (i >= rows * Kout) return;
  const int rr = i / Kout;
  const int k = i - rr * Kout;
  out[i] = (k < Kin) ? f2bf(in[(size_t)rr * Kin + k]) : (unsigned short)0;
}

// ---------------- bf16 GEMM: out[M x Nout] = A @ W^T (+bias)(+relu)(+resid) -----
// one wave per 32x64 output tile; A,W bf16 (K multiple of 32); out fp32 or bf16.
__global__ __launch_bounds__(32)
void gemm_bf16_kernel(const unsigned short* __restrict__ A, int lda,
                      const unsigned short* __restrict__ W, int ldw, int K,
                      const float* __restrict__ bias,
                      const float* __restrict__ resid,
                      float* __restrict__ outf,
                      unsigned short* __restrict__ outh, int ldo, int act) {
  const int mt = blockIdx.x * 32;
  const int ng = blockIdx.y * 64;
  const int l  = threadIdx.x;
  const int r  = l & 15;
  const int offA = (l < 16) ? 0 : 8;
  const int offB = (l < 16) ? 0 : 16;
  v8f acc0[4] = {};
  v8f acc1[4] = {};
  const unsigned short* arow0 = A + (size_t)(mt + r) * lda;
  const unsigned short* arow1 = A + (size_t)(mt + 16 + r) * lda;
  const unsigned short* wrow  = W + (size_t)(ng + r) * ldw;
  for (int kk = 0; kk < K; kk += 32) {
    FragU a0, a1;
    a0.q[0] = *(const u16x8*)(arow0 + kk + offA);
    a0.q[1] = *(const u16x8*)(arow0 + kk + 16 + offA);
    a1.q[0] = *(const u16x8*)(arow1 + kk + offA);
    a1.q[1] = *(const u16x8*)(arow1 + kk + 16 + offA);
#pragma unroll
    for (int g = 0; g < 4; ++g) {
      FragU b;
      const unsigned short* wp = wrow + (size_t)(16 * g) * ldw + kk + offA;
      b.q[0] = *(const u16x8*)(wp);
      b.q[1] = *(const u16x8*)(wp + 16);
      acc0[g] = wmma_bf16(a0, b, acc0[g]);
      acc1[g] = wmma_bf16(a1, b, acc1[g]);
    }
  }
#pragma unroll
  for (int g = 0; g < 4; ++g) {
    const int col = ng + 16 * g + r;
    const float bs = bias[col];
#pragma unroll
    for (int j = 0; j < 8; ++j) {
      const int row0 = mt + j + offA;
      const int row1 = row0 + 16;
      float c0 = acc0[g][j] + bs;
      float c1 = acc1[g][j] + bs;
      if (act) { c0 = fmaxf(c0, 0.0f); c1 = fmaxf(c1, 0.0f); }
      if (resid) {
        c0 += resid[(size_t)row0 * ldo + col];
        c1 += resid[(size_t)row1 * ldo + col];
      }
      if (outh) {
        outh[(size_t)row0 * ldo + col] = f2bf(c0);
        outh[(size_t)row1 * ldo + col] = f2bf(c1);
      } else {
        outf[(size_t)row0 * ldo + col] = c0;
        outf[(size_t)row1 * ldo + col] = c1;
      }
    }
  }
}

// ---------------- Flash-style masked attention ----------------
// one wave per (16 query rows, head).  qkvb: [N][1536] bf16 (Q|K|V),
// Vt: [H][64][N] bf16,  adj: N x N bitmask,  ao: [N][512] bf16.
__global__ __launch_bounds__(32)
void attn_kernel(const unsigned short* __restrict__ qkvb,
                 const unsigned short* __restrict__ Vt,
                 const unsigned int* __restrict__ adj,
                 unsigned short* __restrict__ ao) {
  const int qt = blockIdx.x * 16;
  const int head = blockIdx.y;
  const int l = threadIdx.x;
  const int r = l & 15;
  const int offA = (l < 16) ? 0 : 8;
  const int offB = (l < 16) ? 0 : 16;
  __shared__ __align__(32) unsigned short Pl[16 * 32];

  FragU aq0, aq1;
  {
    const unsigned short* qp = qkvb + (size_t)(qt + r) * QKVD + head * HDIM;
    aq0.q[0] = *(const u16x8*)(qp + offA);
    aq0.q[1] = *(const u16x8*)(qp + 16 + offA);
    aq1.q[0] = *(const u16x8*)(qp + 32 + offA);
    aq1.q[1] = *(const u16x8*)(qp + 48 + offA);
  }
  float m[8], lsum[8];
#pragma unroll
  for (int j = 0; j < 8; ++j) { m[j] = -1e30f; lsum[j] = 0.0f; }
  v8f o[4] = {};
  const unsigned int* adjrow = adj + (size_t)(qt + r) * (NNODE / 32);

  for (int ct = 0; ct < NNODE; ct += 32) {
    unsigned int w = adjrow[ct >> 5];              // lane r holds row qt+r's 32 mask bits
    if (!__any(w != 0u)) continue;                 // fully-masked 16x32 tile -> skip

    FragU bk00, bk01, bk10, bk11;
    const unsigned short* kp0 = qkvb + (size_t)(ct + r) * QKVD + DMODEL + head * HDIM;
    const unsigned short* kp1 = kp0 + (size_t)16 * QKVD;
    bk00.q[0] = *(const u16x8*)(kp0 + offA);
    bk00.q[1] = *(const u16x8*)(kp0 + offA + 16);
    bk01.q[0] = *(const u16x8*)(kp0 + 32 + offA);
    bk01.q[1] = *(const u16x8*)(kp0 + 32 + offA + 16);
    bk10.q[0] = *(const u16x8*)(kp1 + offA);
    bk10.q[1] = *(const u16x8*)(kp1 + offA + 16);
    bk11.q[0] = *(const u16x8*)(kp1 + 32 + offA);
    bk11.q[1] = *(const u16x8*)(kp1 + 32 + offA + 16);

    v8f s0 = {}, s1 = {};
    s0 = wmma_bf16(aq0, bk00, s0);
    s0 = wmma_bf16(aq1, bk01, s0);
    s1 = wmma_bf16(aq0, bk10, s1);
    s1 = wmma_bf16(aq1, bk11, s1);

#pragma unroll
    for (int j = 0; j < 8; ++j) {
      unsigned int wj = (unsigned int)__shfl((int)w, j + offA, 32);
      float v0 = s0[j] * 0.125f;                   // 1/sqrt(64)
      float v1 = s1[j] * 0.125f;
      v0 = ((wj >> r) & 1u) ? v0 : -1e30f;
      v1 = ((wj >> (16 + r)) & 1u) ? v1 : -1e30f;
      float mx = fmaxf(v0, v1);
#pragma unroll
      for (int d2 = 8; d2 >= 1; d2 >>= 1) mx = fmaxf(mx, __shfl_xor(mx, d2, 16));
      float mn = fmaxf(m[j], mx);
      float sc = __expf(m[j] - mn);
      m[j] = mn;
      float p0 = __expf(v0 - mn);
      float p1 = __expf(v1 - mn);
      float rs = p0 + p1;
#pragma unroll
      for (int d2 = 8; d2 >= 1; d2 >>= 1) rs += __shfl_xor(rs, d2, 16);
      lsum[j] = lsum[j] * sc + rs;
      o[0][j] *= sc; o[1][j] *= sc; o[2][j] *= sc; o[3][j] *= sc;
      Pl[(j + offA) * 32 + r]      = f2bf(p0);
      Pl[(j + offA) * 32 + 16 + r] = f2bf(p1);
    }
    __syncthreads();
    FragU pf;  // reload P as A-layout fragment (16x32)
    pf.q[0] = *(const u16x8*)&Pl[r * 32 + offA];
    pf.q[1] = *(const u16x8*)&Pl[r * 32 + 16 + offA];
#pragma unroll
    for (int g = 0; g < 4; ++g) {
      const unsigned short* vp =
          Vt + ((size_t)head * HDIM + 16 * g + r) * NNODE + ct + offA;
      FragU bv;
      bv.q[0] = *(const u16x8*)(vp);
      bv.q[1] = *(const u16x8*)(vp + 16);
      o[g] = wmma_bf16(pf, bv, o[g]);
    }
    __syncthreads();
  }
#pragma unroll
  for (int j = 0; j < 8; ++j) {
    const float inv = 1.0f / lsum[j];              // diagonal guarantees lsum > 0
    const int row = qt + j + offA;
    unsigned short* orow = ao + (size_t)row * DMODEL + head * HDIM;
    orow[r]      = f2bf(o[0][j] * inv);
    orow[16 + r] = f2bf(o[1][j] * inv);
    orow[32 + r] = f2bf(o[2][j] * inv);
    orow[48 + r] = f2bf(o[3][j] * inv);
  }
}

// ---------------- LayerNorm (fp32 in, bf16 out) ----------------
__global__ void ln_kernel(const float* __restrict__ x, const float* __restrict__ g,
                          const float* __restrict__ b, unsigned short* __restrict__ out) {
  const int row = blockIdx.x;
  const int t = threadIdx.x;          // 256 threads, 2 elems each
  const float* xr = x + (size_t)row * DMODEL;
  float v0 = xr[t];
  float v1 = xr[t + 256];
  float s = v0 + v1;
#pragma unroll
  for (int d = 16; d >= 1; d >>= 1) s += __shfl_xor(s, d, 32);
  __shared__ float sm[8], sv[8];
  const int wid = t >> 5;
  if ((t & 31) == 0) sm[wid] = s;
  __syncthreads();
  float tot = 0.0f;
#pragma unroll
  for (int i = 0; i < 8; ++i) tot += sm[i];
  const float mean = tot * (1.0f / DMODEL);
  const float d0 = v0 - mean, d1 = v1 - mean;
  float q = d0 * d0 + d1 * d1;
#pragma unroll
  for (int d = 16; d >= 1; d >>= 1) q += __shfl_xor(q, d, 32);
  if ((t & 31) == 0) sv[wid] = q;
  __syncthreads();
  float vtot = 0.0f;
#pragma unroll
  for (int i = 0; i < 8; ++i) vtot += sv[i];
  const float inv = rsqrtf(vtot * (1.0f / DMODEL) + 1e-5f);
  out[(size_t)row * DMODEL + t]       = f2bf(d0 * inv * g[t] + b[t]);
  out[(size_t)row * DMODEL + t + 256] = f2bf(d1 * inv * g[t + 256] + b[t + 256]);
}

// final LN variant kept fp32->bf16 too (consumed by out GEMM)

// ---------------- embedding / concat (writes bf16, padded to 224) ---------------
__global__ void embed_kernel(const float* __restrict__ x_t, const float* __restrict__ time_t,
                             const int* __restrict__ labels, const float* __restrict__ lap_pe,
                             const float* __restrict__ tw1, const float* __restrict__ tb1,
                             const float* __restrict__ tw2, const float* __restrict__ tb2,
                             const float* __restrict__ cell_embed,
                             unsigned short* __restrict__ comb) {
  const int n = blockIdx.x;
  const int l = threadIdx.x;          // 32 threads
  __shared__ float h1[32];
  const float t = time_t[n];
  float a = t * tw1[l] + tb1[l];
  h1[l] = a > 0.0f ? a : 0.0f;
  __syncthreads();
  float te = tb2[l];
#pragma unroll 4
  for (int j = 0; j < 32; ++j) te += h1[j] * tw2[l * 32 + j];
  const int lab = labels[n];
  unsigned short* cr = comb + (size_t)n * CDIM;
  for (int c = l; c < CDIM; c += 32) {
    float v;
    if (c < 128)       v = x_t[(size_t)n * 128 + c];
    else if (c < 160)  v = te;                       // c == l + 128 here
    else if (c < 192)  v = cell_embed[lab * 32 + (c - 160)];
    else if (c < 208)  v = lap_pe[(size_t)n * 16 + (c - 192)];
    else               v = 0.0f;
    cr[c] = f2bf(v);
  }
}

// ---------------- V transpose (bf16 -> bf16, pure reorder) ----------------
__global__ void prep_v_kernel(const unsigned short* __restrict__ qkvb,
                              unsigned short* __restrict__ Vt) {
  const int i = blockIdx.x * blockDim.x + threadIdx.x;  // over N*512
  const int n = i >> 9;
  const int c = i & 511;
  const int head = c >> 6;
  const int d = c & 63;
  Vt[((size_t)head * HDIM + d) * NNODE + n] =
      qkvb[(size_t)n * QKVD + 2 * DMODEL + head * HDIM + d];
}

// ---------------- adjacency bitmask ----------------
__global__ void scatter_edges_kernel(const int* __restrict__ ei,
                                     unsigned int* __restrict__ adj, int E) {
  const int i = blockIdx.x * blockDim.x + threadIdx.x;
  if (i >= E) return;
  const int rr = ei[i];
  const int cc = ei[E + i];
  atomicOr(&adj[((size_t)rr * NNODE + cc) >> 5], 1u << (cc & 31));
}

__global__ void set_diag_kernel(unsigned int* __restrict__ adj) {
  const int i = blockIdx.x * blockDim.x + threadIdx.x;
  if (i >= NNODE) return;
  atomicOr(&adj[((size_t)i * NNODE + i) >> 5], 1u << (i & 31));
}

extern "C" void kernel_launch(void* const* d_in, const int* in_sizes, int n_in,
                              void* d_out, int out_size, void* d_ws, size_t ws_size,
                              hipStream_t stream) {
  // setup_inputs() insertion order (params dict in insertion order, layers expanded)
  const float* x_t        = (const float*)d_in[0];
  const float* time_t     = (const float*)d_in[1];
  const int*   labels     = (const int*)d_in[2];
  const int*   edges      = (const int*)d_in[3];
  const float* lap_pe     = (const float*)d_in[4];
  const float* time_w1    = (const float*)d_in[5];
  const float* time_b1    = (const float*)d_in[6];
  const float* time_w2    = (const float*)d_in[7];
  const float* time_b2    = (const float*)d_in[8];
  const float* cell_embed = (const float*)d_in[9];
  const float* proj_w     = (const float*)d_in[10];
  const float* proj_b     = (const float*)d_in[11];
  const float* fn_g       = (const float*)d_in[12];
  const float* fn_b       = (const float*)d_in[13];
  const float* out_w      = (const float*)d_in[14];
  const float* out_b      = (const float*)d_in[15];
  auto lp = [&](int layer, int idx) { return (const float*)d_in[16 + layer * 12 + idx]; };
  const int E = in_sizes[3] / 2;

  char* wp = (char*)d_ws;
  auto take = [&](size_t bytes) {
    char* p = wp;
    wp += (bytes + 255) & ~(size_t)255;
    return p;
  };
  unsigned int* adj    = (unsigned int*)take((size_t)NNODE * NNODE / 8);   // 2 MB bitmask
  unsigned short* comb = (unsigned short*)take((size_t)NNODE * CDIM * 2);
  float* h             = (float*)take((size_t)NNODE * DMODEL * 4);
  unsigned short* xlnb = (unsigned short*)take((size_t)NNODE * DMODEL * 2);
  unsigned short* qkvb = (unsigned short*)take((size_t)NNODE * QKVD * 2);
  unsigned short* Vt   = (unsigned short*)take((size_t)NNODE * DMODEL * 2);
  unsigned short* aob  = (unsigned short*)take((size_t)NNODE * DMODEL * 2);
  unsigned short* mlp1b = (unsigned short*)take((size_t)NNODE * 4 * DMODEL * 2);
  // bf16 weights
  unsigned short* projw_b = (unsigned short*)take((size_t)DMODEL * CDIM * 2);
  unsigned short* outw_b  = (unsigned short*)take((size_t)128 * DMODEL * 2);
  unsigned short* wi_b[2], *wo_b[2], *w1_b[2], *w2_b[2];
  for (int i = 0; i < 2; ++i) {
    wi_b[i] = (unsigned short*)take((size_t)3 * DMODEL * DMODEL * 2);
    wo_b[i] = (unsigned short*)take((size_t)DMODEL * DMODEL * 2);
    w1_b[i] = (unsigned short*)take((size_t)4 * DMODEL * DMODEL * 2);
    w2_b[i] = (unsigned short*)take((size_t)DMODEL * 4 * DMODEL * 2);
  }

  // ---- adjacency bitmask ----
  hipMemsetAsync(adj, 0, (size_t)NNODE * NNODE / 8, stream);
  scatter_edges_kernel<<<(E + 255) / 256, 256, 0, stream>>>(edges, adj, E);
  set_diag_kernel<<<(NNODE + 255) / 256, 256, 0, stream>>>(adj);

  // ---- one-time weight conversion to bf16 ----
  auto conv = [&](const float* src, unsigned short* dst, int n) {
    conv_bf16_kernel<<<(n + 255) / 256, 256, 0, stream>>>(src, dst, n);
  };
  conv_pad_kernel<<<(DMODEL * CDIM + 255) / 256, 256, 0, stream>>>(
      proj_w, projw_b, DMODEL, CREAL, CDIM);
  conv(out_w, outw_b, 128 * DMODEL);
  for (int i = 0; i < 2; ++i) {
    conv(lp(i, 2),  wi_b[i], 3 * DMODEL * DMODEL);
    conv(lp(i, 4),  wo_b[i], DMODEL * DMODEL);
    conv(lp(i, 8),  w1_b[i], 4 * DMODEL * DMODEL);
    conv(lp(i, 10), w2_b[i], DMODEL * 4 * DMODEL);
  }

  // ---- embedding + input projection ----
  embed_kernel<<<NNODE, 32, 0, stream>>>(x_t, time_t, labels, lap_pe, time_w1,
                                         time_b1, time_w2, time_b2, cell_embed, comb);
  gemm_bf16_kernel<<<dim3(NNODE / 32, DMODEL / 64), 32, 0, stream>>>(
      comb, CDIM, projw_b, CDIM, CDIM, proj_b, nullptr, h, nullptr, DMODEL, 0);

  for (int i = 0; i < 2; ++i) {
    const float *ln1_g = lp(i, 0), *ln1_b = lp(i, 1), *bi = lp(i, 3);
    const float *bo = lp(i, 5), *ln2_g = lp(i, 6), *ln2_b = lp(i, 7);
    const float *b1 = lp(i, 9), *b2 = lp(i, 11);

    ln_kernel<<<NNODE, 256, 0, stream>>>(h, ln1_g, ln1_b, xlnb);
    gemm_bf16_kernel<<<dim3(NNODE / 32, QKVD / 64), 32, 0, stream>>>(
        xlnb, DMODEL, wi_b[i], DMODEL, DMODEL, bi, nullptr, nullptr, qkvb, QKVD, 0);
    prep_v_kernel<<<(NNODE * DMODEL) / 256, 256, 0, stream>>>(qkvb, Vt);
    attn_kernel<<<dim3(NNODE / 16, NHEAD), 32, 0, stream>>>(qkvb, Vt, adj, aob);
    // h = h + ao @ wo^T + bo
    gemm_bf16_kernel<<<dim3(NNODE / 32, DMODEL / 64), 32, 0, stream>>>(
        aob, DMODEL, wo_b[i], DMODEL, DMODEL, bo, h, h, nullptr, DMODEL, 0);
    ln_kernel<<<NNODE, 256, 0, stream>>>(h, ln2_g, ln2_b, xlnb);
    gemm_bf16_kernel<<<dim3(NNODE / 32, 4 * DMODEL / 64), 32, 0, stream>>>(
        xlnb, DMODEL, w1_b[i], DMODEL, DMODEL, b1, nullptr, nullptr, mlp1b,
        4 * DMODEL, 1);
    // h = h + relu(...) @ w2^T + b2
    gemm_bf16_kernel<<<dim3(NNODE / 32, DMODEL / 64), 32, 0, stream>>>(
        mlp1b, 4 * DMODEL, w2_b[i], 4 * DMODEL, 4 * DMODEL, b2, h, h, nullptr,
        DMODEL, 0);
  }

  ln_kernel<<<NNODE, 256, 0, stream>>>(h, fn_g, fn_b, xlnb);
  gemm_bf16_kernel<<<dim3(NNODE / 32, 128 / 64), 32, 0, stream>>>(
      xlnb, DMODEL, outw_b, DMODEL, DMODEL, out_b, nullptr, (float*)d_out, nullptr,
      128, 0);
}
